// MHeadQKV_34806414967368
// MI455X (gfx1250) — matrix operand order repaired and checked by reading, not compiled, not run on hardware
//
#include <hip/hip_runtime.h>
#include <hip/hip_bf16.h>

// ---------------------------------------------------------------------------
// MHeadQKV (retention attention) for MI455X / gfx1250, bf16 WMMA pipeline.
//   1) cast X -> bf16, cast+transpose W_{Q,K,V} -> bf16 (N-major)
//   2) proj GEMM (X @ W) with fused RoPE + SiLU epilogue
//        Q,K stored [H][T][C] row-major; V stored transposed [H][C][T]
//   3) Ahat[h][k][q] = (K Q^T) * gamma^(q-k) * (q>=k), bf16, triangular skip
//   4) Osum[t][c]    = sum_h  Ahat[h] @ V[h]   (heads accumulated in regs)
//   5) GroupNorm(16 groups) -> d_out (fp32)
// All matrix math via v_wmma_f32_16x16x32_bf16 (fp32 accumulate).
// ---------------------------------------------------------------------------

typedef __bf16 bf16_t;
typedef __attribute__((ext_vector_type(16))) __bf16      v16bf;
typedef __attribute__((ext_vector_type(8)))  float       v8f;
typedef __attribute__((ext_vector_type(4)))  unsigned int v4u;

#define T_DIM 2048
#define C_DIM 1024
#define H_DIM 8
#define N_PROJ (C_DIM * H_DIM) /* 8192 */

// A-matrix 16x32 bf16 fragment (row-major source, contiguous along K).
// Lanes 0-15: row M=lane, K = k0+{0..7}, k0+{16..23}
// Lanes 16-31: row M=lane-16, K = k0+{8..15}, k0+{24..31}
__device__ __forceinline__ v16bf load_a_frag(const bf16_t* __restrict__ base,
                                             int row0, int k0, int pitch, int lane) {
  const int l  = lane & 15;
  const int hi = (lane >> 4) * 8;
  const bf16_t* p = base + (size_t)(row0 + l) * (size_t)pitch + (k0 + hi);
  union { v16bf v; v4u u[2]; } f;
  f.u[0] = *reinterpret_cast<const v4u*>(p);
  f.u[1] = *reinterpret_cast<const v4u*>(p + 16);
  return f.v;
}

// B-matrix 32x16 bf16 fragment from an N-major (transposed) source:
// lane holds column N = col0 + (lane&15); lanes 0-15 K=k0+{0..7,16..23}, lanes 16-31 K=k0+{8..15,24..31}.
__device__ __forceinline__ v16bf load_b_frag(const bf16_t* __restrict__ baseT,
                                             int col0, int k0, int pitch, int lane) {
  const int l  = lane & 15;
  const int hi = (lane >> 4) * 8;
  const bf16_t* p = baseT + (size_t)(col0 + l) * (size_t)pitch + (k0 + hi);
  union { v16bf v; v4u u[2]; } f;
  f.u[0] = *reinterpret_cast<const v4u*>(p);
  f.u[1] = *reinterpret_cast<const v4u*>(p + 16);
  return f.v;
}

__device__ __forceinline__ v8f wmma_bf16(v16bf a, v16bf b, v8f c) {
  return __builtin_amdgcn_wmma_f32_16x16x32_bf16(false, a, false, b, (short)0, c,
                                                 false, false);
}

__device__ __forceinline__ float silu(float x) {
  return x / (1.f + __expf(-x));
}

// ---------------------------------------------------------------------------
// Stage 1: casts
// ---------------------------------------------------------------------------
__global__ void cast_x_kernel(const float* __restrict__ X, bf16_t* __restrict__ Xb, int n) {
  int i = blockIdx.x * blockDim.x + threadIdx.x;
  if (i < n) Xb[i] = (bf16_t)X[i];
}

// W: (C_DIM, N_PROJ) row-major -> Wt: (N_PROJ, C_DIM) bf16
__global__ void cast_w_transpose_kernel(const float* __restrict__ W, bf16_t* __restrict__ Wt) {
  int i = blockIdx.x * blockDim.x + threadIdx.x;
  if (i >= C_DIM * N_PROJ) return;
  int c = i / N_PROJ;
  int n = i - c * N_PROJ;
  Wt[(size_t)n * C_DIM + c] = (bf16_t)W[i];
}

// ---------------------------------------------------------------------------
// Stage 2: projection GEMM + fused RoPE + SiLU
// Block 128 thr (4 waves, 2x2), block tile 128x64, wave tile 64x32 (4x2 WMMA).
// mode 0: Q (rope+silu, [H][T][C]); 1: K (same); 2: V (silu, [H][C][T]).
// ---------------------------------------------------------------------------
__global__ void proj_rope_silu_kernel(const bf16_t* __restrict__ Xb,
                                      const bf16_t* __restrict__ Wt,
                                      bf16_t* __restrict__ out, int mode) {
  const int lane = threadIdx.x & 31;
  const int wave = threadIdx.x >> 5;
  const int m0 = blockIdx.y * 128 + (wave >> 1) * 64;
  const int n0 = blockIdx.x * 64 + (wave & 1) * 32;

  const v8f z = {0.f, 0.f, 0.f, 0.f, 0.f, 0.f, 0.f, 0.f};
  v8f acc[4][2];
#pragma unroll
  for (int i = 0; i < 4; ++i)
#pragma unroll
    for (int j = 0; j < 2; ++j) acc[i][j] = z;

  for (int k = 0; k < C_DIM; k += 32) {
    // Speculative prefetch of next K-tile (harmlessly dropped past the end).
    __builtin_prefetch(Xb + (size_t)(m0 + (lane & 15)) * C_DIM + k + 32, 0, 1);
    __builtin_prefetch(Wt + (size_t)(n0 + (lane & 15)) * C_DIM + k + 32, 0, 1);
    v16bf a[4], b[2];
#pragma unroll
    for (int i = 0; i < 4; ++i) a[i] = load_a_frag(Xb, m0 + 16 * i, k, C_DIM, lane);
#pragma unroll
    for (int j = 0; j < 2; ++j) b[j] = load_b_frag(Wt, n0 + 16 * j, k, C_DIM, lane);
#pragma unroll
    for (int i = 0; i < 4; ++i)
#pragma unroll
      for (int j = 0; j < 2; ++j) acc[i][j] = wmma_bf16(a[i], b[j], acc[i][j]);
  }

  const float kInvFreqScale = 0.14391156516f;  // ln(10000)/64
  const int mrow = (lane < 16) ? 0 : 8;
  const int ncol = lane & 15;
#pragma unroll
  for (int j = 0; j < 2; ++j) {
    const int n = n0 + 16 * j + ncol;
    const int h = n >> 10;
    const int c = n & (C_DIM - 1);
    // Wave-uniform: this 16-col tile touches RoPE channels iff its base ch < 64.
    const bool tileRope = (mode < 2) && (((n0 + 16 * j) & (C_DIM - 1)) < 64);
#pragma unroll
    for (int i = 0; i < 4; ++i) {
#pragma unroll
      for (int e = 0; e < 8; ++e) {
        const int m = m0 + 16 * i + mrow + e;  // t
        float v = acc[i][j][e];
        if (tileRope) {
          // RoPE pairs (c, c^1) live on adjacent lanes in the C layout.
          const float partner = __shfl_xor(v, 1, 32);
          const float ang = (float)m * __expf(-(float)(c & ~1) * kInvFreqScale);
          float sn, cs;
          __sincosf(ang, &sn, &cs);
          v = (c & 1) ? (v * cs + partner * sn) : (v * cs - partner * sn);
        }
        v = silu(v);
        const bf16_t r = (bf16_t)v;
        if (mode == 2)
          out[((size_t)h * C_DIM + c) * T_DIM + m] = r;  // V transposed [H][C][T]
        else
          out[((size_t)h * T_DIM + m) * C_DIM + c] = r;  // [H][T][C]
      }
    }
  }
}

// ---------------------------------------------------------------------------
// Stage 3: Ahat[h][k][q] = (K[h] @ Q[h]^T) * gamma^(q-k) for q>=k else 0 (bf16)
// Blocks with q_hi < k_lo are never read downstream -> skipped.
// ---------------------------------------------------------------------------
__global__ void attn_qk_decay_kernel(const bf16_t* __restrict__ Kb,
                                     const bf16_t* __restrict__ Qb,
                                     bf16_t* __restrict__ Ahat) {
  const int h = blockIdx.z;
  const int kb = blockIdx.y * 128;
  const int qb = blockIdx.x * 64;
  if (qb + 64 <= kb) return;  // strictly-zero block, never consumed

  const int lane = threadIdx.x & 31;
  const int wave = threadIdx.x >> 5;
  const int m0 = kb + (wave >> 1) * 64;  // k rows
  const int n0 = qb + (wave & 1) * 32;   // q cols

  const bf16_t* A = Kb + (size_t)h * T_DIM * C_DIM;
  const bf16_t* B = Qb + (size_t)h * T_DIM * C_DIM;

  const v8f z = {0.f, 0.f, 0.f, 0.f, 0.f, 0.f, 0.f, 0.f};
  v8f acc[4][2];
#pragma unroll
  for (int i = 0; i < 4; ++i)
#pragma unroll
    for (int j = 0; j < 2; ++j) acc[i][j] = z;

  for (int k = 0; k < C_DIM; k += 32) {
    __builtin_prefetch(A + (size_t)(m0 + (lane & 15)) * C_DIM + k + 32, 0, 1);
    __builtin_prefetch(B + (size_t)(n0 + (lane & 15)) * C_DIM + k + 32, 0, 1);
    v16bf a[4], b[2];
#pragma unroll
    for (int i = 0; i < 4; ++i) a[i] = load_a_frag(A, m0 + 16 * i, k, C_DIM, lane);
#pragma unroll
    for (int j = 0; j < 2; ++j) b[j] = load_b_frag(B, n0 + 16 * j, k, C_DIM, lane);
#pragma unroll
    for (int i = 0; i < 4; ++i)
#pragma unroll
      for (int j = 0; j < 2; ++j) acc[i][j] = wmma_bf16(a[i], b[j], acc[i][j]);
  }

  const float kLnGamma = -0.03174869831f;  // ln(0.96875)
  bf16_t* dst = Ahat + (size_t)h * T_DIM * T_DIM;
  const int mrow = (lane < 16) ? 0 : 8;
  const int ncol = lane & 15;
#pragma unroll
  for (int i = 0; i < 4; ++i)
#pragma unroll
    for (int j = 0; j < 2; ++j)
#pragma unroll
      for (int e = 0; e < 8; ++e) {
        const int m = m0 + 16 * i + mrow + e;  // k
        const int n = n0 + 16 * j + ncol;      // q
        const int diff = n - m;                // q - k
        float v = acc[i][j][e];
        v = (diff >= 0) ? v * __expf((float)diff * kLnGamma) : 0.f;
        dst[(size_t)m * T_DIM + n] = (bf16_t)v;
      }
}

// ---------------------------------------------------------------------------
// Stage 4: Osum[k][c] = sum_h sum_{q>=k-block} Ahat[h][k][q] * V[h][q][c]
// Heads accumulated in registers; q loop starts at the 128-aligned k block.
// ---------------------------------------------------------------------------
__global__ void attn_av_sum_kernel(const bf16_t* __restrict__ Ahat,
                                   const bf16_t* __restrict__ Vt,
                                   float* __restrict__ Osum) {
  const int kb = blockIdx.y * 128;
  const int cb = blockIdx.x * 64;
  const int lane = threadIdx.x & 31;
  const int wave = threadIdx.x >> 5;
  const int m0 = kb + (wave >> 1) * 64;
  const int n0 = cb + (wave & 1) * 32;

  const v8f z = {0.f, 0.f, 0.f, 0.f, 0.f, 0.f, 0.f, 0.f};
  v8f acc[4][2];
#pragma unroll
  for (int i = 0; i < 4; ++i)
#pragma unroll
    for (int j = 0; j < 2; ++j) acc[i][j] = z;

  for (int h = 0; h < H_DIM; ++h) {
    const bf16_t* A = Ahat + (size_t)h * T_DIM * T_DIM;  // [k][q], pitch T
    const bf16_t* B = Vt + (size_t)h * C_DIM * T_DIM;    // [c][q], pitch T
    for (int q = kb; q < T_DIM; q += 32) {
      __builtin_prefetch(A + (size_t)(m0 + (lane & 15)) * T_DIM + q + 32, 0, 1);
      __builtin_prefetch(B + (size_t)(n0 + (lane & 15)) * T_DIM + q + 32, 0, 1);
      v16bf a[4], b[2];
#pragma unroll
      for (int i = 0; i < 4; ++i) a[i] = load_a_frag(A, m0 + 16 * i, q, T_DIM, lane);
#pragma unroll
      for (int j = 0; j < 2; ++j) b[j] = load_b_frag(B, n0 + 16 * j, q, T_DIM, lane);
#pragma unroll
      for (int i = 0; i < 4; ++i)
#pragma unroll
        for (int j = 0; j < 2; ++j) acc[i][j] = wmma_bf16(a[i], b[j], acc[i][j]);
    }
  }

  const int mrow = (lane < 16) ? 0 : 8;
  const int ncol = lane & 15;
#pragma unroll
  for (int i = 0; i < 4; ++i)
#pragma unroll
    for (int j = 0; j < 2; ++j)
#pragma unroll
      for (int e = 0; e < 8; ++e) {
        const int m = m0 + 16 * i + mrow + e;
        const int n = n0 + 16 * j + ncol;
        Osum[(size_t)m * C_DIM + n] = acc[i][j][e];
      }
}

// ---------------------------------------------------------------------------
// Stage 5: GroupNorm, 16 groups of 64 channels. One block (256 thr) per row t.
// ---------------------------------------------------------------------------
__global__ void groupnorm_kernel(const float* __restrict__ Osum,
                                 const float* __restrict__ w,
                                 const float* __restrict__ b,
                                 float* __restrict__ out) {
  __shared__ float s_sum[256];
  __shared__ float s_sq[256];
  const int t = blockIdx.x;
  const int tid = threadIdx.x;
  const int c0 = tid * 4;
  const float4 x = *reinterpret_cast<const float4*>(Osum + (size_t)t * C_DIM + c0);
  s_sum[tid] = x.x + x.y + x.z + x.w;
  s_sq[tid] = x.x * x.x + x.y * x.y + x.z * x.z + x.w * x.w;
  __syncthreads();
  const int g = tid >> 4;  // 16 threads/group * 4 ch = 64 ch/group
  if ((tid & 15) == 0) {
    float ss = 0.f, qq = 0.f;
#pragma unroll
    for (int k = 0; k < 16; ++k) { ss += s_sum[tid + k]; qq += s_sq[tid + k]; }
    s_sum[tid] = ss; s_sq[tid] = qq;
  }
  __syncthreads();
  const float mu = s_sum[g << 4] * (1.f / 64.f);
  const float var = s_sq[g << 4] * (1.f / 64.f) - mu * mu;
  const float rstd = rsqrtf(var + 1e-5f);
  float4 o;
  o.x = (x.x - mu) * rstd * w[c0 + 0] + b[c0 + 0];
  o.y = (x.y - mu) * rstd * w[c0 + 1] + b[c0 + 1];
  o.z = (x.z - mu) * rstd * w[c0 + 2] + b[c0 + 2];
  o.w = (x.w - mu) * rstd * w[c0 + 3] + b[c0 + 3];
  *reinterpret_cast<float4*>(out + (size_t)t * C_DIM + c0) = o;
}

// ---------------------------------------------------------------------------
extern "C" void kernel_launch(void* const* d_in, const int* in_sizes, int n_in,
                              void* d_out, int out_size, void* d_ws, size_t ws_size,
                              hipStream_t stream) {
  const float* X  = (const float*)d_in[0];
  const float* WQ = (const float*)d_in[1];
  const float* WK = (const float*)d_in[2];
  const float* WV = (const float*)d_in[3];
  const float* gw = (const float*)d_in[4];
  const float* gb = (const float*)d_in[5];
  float* out = (float*)d_out;

  char* ws = (char*)d_ws;
  const size_t MB = 1024ull * 1024ull;
  // Peak workspace ~168 MB. Ahat reuses the Xb/Wt staging region (dead by then).
  bf16_t* Qb   = (bf16_t*)(ws + 0 * MB);    // 32 MB  [H][T][C]
  bf16_t* Kb   = (bf16_t*)(ws + 32 * MB);   // 32 MB  [H][T][C]
  bf16_t* Vt   = (bf16_t*)(ws + 64 * MB);   // 32 MB  [H][C][T]
  bf16_t* Xb   = (bf16_t*)(ws + 96 * MB);   //  4 MB
  bf16_t* WtQ  = (bf16_t*)(ws + 100 * MB);  // 16 MB
  bf16_t* WtK  = (bf16_t*)(ws + 116 * MB);  // 16 MB
  bf16_t* WtV  = (bf16_t*)(ws + 132 * MB);  // 16 MB
  bf16_t* Ahat = (bf16_t*)(ws + 96 * MB);   // 64 MB  [H][T][T] (reuse)
  float*  Os   = (float*)(ws + 160 * MB);   //  8 MB  [T][C]

  cast_x_kernel<<<(T_DIM * C_DIM) / 256, 256, 0, stream>>>(X, Xb, T_DIM * C_DIM);
  const int wblocks = (C_DIM * N_PROJ) / 256;
  cast_w_transpose_kernel<<<wblocks, 256, 0, stream>>>(WQ, WtQ);
  cast_w_transpose_kernel<<<wblocks, 256, 0, stream>>>(WK, WtK);
  cast_w_transpose_kernel<<<wblocks, 256, 0, stream>>>(WV, WtV);

  dim3 gp(N_PROJ / 64, T_DIM / 128);
  proj_rope_silu_kernel<<<gp, 128, 0, stream>>>(Xb, WtQ, Qb, 0);
  proj_rope_silu_kernel<<<gp, 128, 0, stream>>>(Xb, WtK, Kb, 1);
  proj_rope_silu_kernel<<<gp, 128, 0, stream>>>(Xb, WtV, Vt, 2);

  dim3 g1(T_DIM / 64, T_DIM / 128, H_DIM);
  attn_qk_decay_kernel<<<g1, 128, 0, stream>>>(Kb, Qb, Ahat);

  dim3 g2(C_DIM / 64, T_DIM / 128);
  attn_av_sum_kernel<<<g2, 128, 0, stream>>>(Ahat, Vt, Os);

  groupnorm_kernel<<<T_DIM, 256, 0, stream>>>(Os, gw, gb, out);
}
